// MyModel_39676907882096
// MI455X (gfx1250) — compile-verified, matrix operand order checked by reading, not run on hardware
//
#include <hip/hip_runtime.h>
#include <hip/hip_bf16.h>
#include <math.h>

// ---------------------------------------------------------------------------
// Flash attention for B=8, Sq=Sk=2048, D=128, fp32 in/out, multiplicative
// mask on scores before softmax (DIM=1 -> q-scale is exactly 1.0).
//
// MI455X reasoning:
//  * mask is 128 MB and must stream exactly once -> HBM floor ~7us @23.3TB/s.
//    Online softmax keeps us at that floor; mask/out use non-temporal hints.
//  * K/V tiles shared by the block's 4 waves through LDS, staged with CDNA5
//    GLOBAL_LOAD_ASYNC_TO_LDS_B128 (ASYNCcnt), double-buffered so the async
//    engine fills tile i+1 while the WMMAs chew tile i.
//  * Softmax row-max via DPP16 butterflies (VALU, dual-issue) instead of
//    ds_bpermute; row-sum reduction deferred to the epilogue (per-lane
//    partials are exact because corr is lane-uniform). Keeps the DS unit
//    free to feed WMMA fragments.
//  * Compute: v_wmma_f32_16x16x32_f16, 16 per 32-key iteration, fragment
//    loads software-pipelined one step ahead of the WMMA chain.
//
// Workspace layout (12 MB, f16):
//   qh  [B][Sq][D]   row-major f16 copy of q
//   kh  [B][Sk][D]   row-major f16 copy of k
//   vt  [B][D][Sk]   f16 transpose of v (so B-fragments of V are contiguous)
// ---------------------------------------------------------------------------

typedef _Float16 v16h __attribute__((ext_vector_type(16)));
typedef _Float16 v8h  __attribute__((ext_vector_type(8)));
typedef float    v8f  __attribute__((ext_vector_type(8)));

constexpr int B  = 8;
constexpr int SQ = 2048;
constexpr int SK = 2048;
constexpr int D  = 128;
constexpr int NIT = SK / 32;   // 64 key-tile iterations

// ---------------------------------------------------------------------------
// CDNA5 async copy: 16 bytes/lane global -> LDS, tracked by ASYNCcnt.
// VDST holds the per-lane LDS byte offset (low 32 bits of a generic LDS
// pointer are the LDS offset on CDNA5: aperture rules, CDNA5 ISA).
// ---------------------------------------------------------------------------
__device__ inline void async_copy16(const _Float16* g, _Float16* l) {
  uint32_t loff = (uint32_t)(uintptr_t)l;
  asm volatile("global_load_async_to_lds_b128 %0, %1, off"
               :: "v"(loff), "v"(g) : "memory");
}

template <int N> __device__ inline void wait_asynccnt() {
#if __has_builtin(__builtin_amdgcn_s_wait_asynccnt)
  __builtin_amdgcn_s_wait_asynccnt(N);
#else
  asm volatile("s_wait_asynccnt %0" :: "i"(N) : "memory");
#endif
}

// ---------------------------------------------------------------------------
// DPP16 butterfly reductions over each 16-lane half (rows live per-half in
// the WMMA C/D layout).  Steps: quad_perm(1,0,3,2)=0xB1, quad_perm(2,3,0,1)
// =0x4E, row_half_mirror=0x141, row_mirror=0x140.  Pure VALU, no LDS.
// ---------------------------------------------------------------------------
template <int CTRL> __device__ inline float dpp_mov(float v) {
  return __int_as_float(__builtin_amdgcn_update_dpp(
      0, __float_as_int(v), CTRL, 0xf, 0xf, true));
}
__device__ inline float red_max16(float v) {
  v = fmaxf(v, dpp_mov<0xB1>(v));
  v = fmaxf(v, dpp_mov<0x4E>(v));
  v = fmaxf(v, dpp_mov<0x141>(v));
  v = fmaxf(v, dpp_mov<0x140>(v));
  return v;
}
__device__ inline float red_sum16(float v) {
  v += dpp_mov<0xB1>(v);
  v += dpp_mov<0x4E>(v);
  v += dpp_mov<0x141>(v);
  v += dpp_mov<0x140>(v);
  return v;
}

// ---------------------------------------------------------------------------
// Fragment loader for the CDNA5 16-bit 16x32 A layout (and, symmetrically,
// the 32x16 B layout, where lane n holds column n = a row of the row-major
// source):  lanes 0-15 : elems 0-7 -> K = k0+0..7,  elems 8-15 -> K = k0+16..23
//           lanes 16-31: elems 0-7 -> K = k0+8..15, elems 8-15 -> K = k0+24..31
// Two 16-byte loads per fragment; all offsets are 16B-aligned. Works for both
// global (b128) and LDS (ds_load_b128) sources.
// ---------------------------------------------------------------------------
__device__ inline v16h load_frag16(const _Float16* base, int stride, int row,
                                   int k0, int half) {
  const _Float16* p = base + (size_t)row * stride + k0 + 8 * half;
  v8h lo = *(const v8h*)(p);
  v8h hi = *(const v8h*)(p + 16);
  return __builtin_shufflevector(lo, hi, 0, 1, 2, 3, 4, 5, 6, 7,
                                         8, 9, 10, 11, 12, 13, 14, 15);
}

// ---------------------------------------------------------------------------
// Prep kernel 1: elementwise f32 -> f16 for q and k.
// ---------------------------------------------------------------------------
__global__ void convert_qk_kernel(const float* __restrict__ q,
                                  const float* __restrict__ k,
                                  _Float16* __restrict__ qh,
                                  _Float16* __restrict__ kh, int n) {
  int i = blockIdx.x * blockDim.x + threadIdx.x;
  if (i < n) {
    qh[i] = (_Float16)q[i];   // DIM=1 -> scale factor DIM**0.5 == 1.0
    kh[i] = (_Float16)k[i];
  }
}

// ---------------------------------------------------------------------------
// Prep kernel 2: LDS-tiled transpose v[B][Sk][D] f32 -> vt[B][D][Sk] f16.
// ---------------------------------------------------------------------------
__global__ void transpose_v_kernel(const float* __restrict__ v,
                                   _Float16* __restrict__ vt) {
  __shared__ _Float16 tile[32][33];
  int b  = blockIdx.z;
  int s0 = blockIdx.x * 32;   // Sk tile base
  int d0 = blockIdx.y * 32;   // D tile base
  int tx = threadIdx.x;       // 0..31
  int ty = threadIdx.y;       // 0..7
#pragma unroll
  for (int i = 0; i < 32; i += 8)
    tile[ty + i][tx] = (_Float16)v[((size_t)b * SK + s0 + ty + i) * D + d0 + tx];
  __syncthreads();
#pragma unroll
  for (int i = 0; i < 32; i += 8)
    vt[((size_t)b * D + d0 + ty + i) * SK + s0 + tx] = tile[tx][ty + i];
}

// ---------------------------------------------------------------------------
// Flash-attention kernel. 4 waves / block; each wave owns one 16-query tile;
// all 4 waves share the double-buffered K/V LDS tiles.
// Grid = B * (SQ/16) / 4 = 256 blocks -> ~1 wave per SIMD32 on a big chip.
// ---------------------------------------------------------------------------
__global__ __launch_bounds__(128) void flash_attn_kernel(
    const float* __restrict__ mask, const _Float16* __restrict__ qh,
    const _Float16* __restrict__ kh, const _Float16* __restrict__ vt,
    float* __restrict__ out) {
  // Double-buffered shared tiles: K tile [32 keys][128 d], V^T tile
  // [128 d][32 keys].  8 KB each per buffer -> 32 KB, + 4 KB P staging.
  __shared__ _Float16 kbuf[2][32][128];
  __shared__ _Float16 vbuf[2][128][32];
  __shared__ _Float16 p_lds[4][16][32];

  const int tid  = threadIdx.x;
  const int wave = tid >> 5;
  const int lane = tid & 31;
  const int half = lane >> 4;   // 0: lanes 0-15, 1: lanes 16-31
  const int col  = lane & 15;

  const int gw = blockIdx.x * 4 + wave;      // global 16-query tile id
  const int b  = gw >> 7;                    // / (SQ/16)
  const int q0 = (gw & 127) << 4;

  const _Float16* qbase = qh + (size_t)b * SQ * D;
  const _Float16* kbase = kh + (size_t)b * SK * D;
  const _Float16* vbase = vt + (size_t)b * D * SK;
  const float*    mbase = mask + ((size_t)b * SQ + q0) * SK;

  // Cooperative async staging of one 32-key iteration's K and V^T tiles:
  // 1024 x 16B chunks over 128 threads -> 8 async b128 per thread.
  auto stage = [&](int buf, int kk0) {
#pragma unroll
    for (int i = 0; i < 4; ++i) {           // K: 512 chunks, 16 per key row
      int c = tid + 128 * i;
      async_copy16(kbase + (size_t)(kk0 + (c >> 4)) * D + (c & 15) * 8,
                   &kbuf[buf][c >> 4][(c & 15) * 8]);
    }
#pragma unroll
    for (int i = 0; i < 4; ++i) {           // V^T: 512 chunks, 4 per d row
      int c = tid + 128 * i;
      async_copy16(vbase + (size_t)(c >> 2) * SK + kk0 + (c & 3) * 8,
                   &vbuf[buf][c >> 2][(c & 3) * 8]);
    }
  };

  // Q A-fragments: 4 chunks of 32 along D (resident for the whole kernel).
  v16h aq[4];
#pragma unroll
  for (int c = 0; c < 4; ++c)
    aq[c] = load_frag16(qbase, D, q0 + col, 32 * c, half);

  // Online-softmax state. C/D layout: VGPR r of this lane holds row
  // M = r + 8*half, column N = col.  lsum holds PER-LANE partial
  // denominators (reduced across the 16-lane half only in the epilogue).
  v8f  acc[8];
  v8f  zero = {};
  float mrow[8], lsum[8];
#pragma unroll
  for (int j = 0; j < 8; ++j) acc[j] = zero;
#pragma unroll
  for (int r = 0; r < 8; ++r) { mrow[r] = -INFINITY; lsum[r] = 0.f; }

  stage(0, 0);   // prologue fill of buffer 0

  for (int it = 0; it < NIT; ++it) {
    const int cur = it & 1;
    const int kk0 = it * 32;
    // Kick off next tile into the other buffer, then drain current tile:
    // after issuing, outstanding = 8 (cur) + 8 (next); wait <=8 -> cur done.
    if (it + 1 < NIT) {
      stage(cur ^ 1, kk0 + 32);
      wait_asynccnt<8>();
    } else {
      wait_asynccnt<0>();
    }
    __syncthreads();   // everyone's share of the current tile is in LDS

    // ---- S = Q K^T from the LDS K tile: two 16-key score tiles, with the
    //      fragment loads pipelined one D-chunk ahead of the WMMA chain. ----
    const _Float16* kb = &kbuf[cur][0][0];
    v16h bk0 = load_frag16(kb, D, col, 0, half);
    v16h bk1 = load_frag16(kb, D, 16 + col, 0, half);
    v8f s0 = zero, s1 = zero;
#pragma unroll
    for (int c = 0; c < 4; ++c) {
      v16h n0, n1;
      if (c < 3) {
        n0 = load_frag16(kb, D, col, 32 * (c + 1), half);
        n1 = load_frag16(kb, D, 16 + col, 32 * (c + 1), half);
      }
      s0 = __builtin_amdgcn_wmma_f32_16x16x32_f16(
          false, aq[c], false, bk0, (short)0, s0, false, false);
      s1 = __builtin_amdgcn_wmma_f32_16x16x32_f16(
          false, aq[c], false, bk1, (short)0, s1, false, false);
      if (c < 3) { bk0 = n0; bk1 = n1; }
    }

    // Multiplicative mask BEFORE softmax; streaming -> non-temporal.
#pragma unroll
    for (int r = 0; r < 8; ++r) {
      const float* mrowp = &mbase[(size_t)(r + 8 * half) * SK + kk0 + col];
      s0[r] *= __builtin_nontemporal_load(mrowp);
      s1[r] *= __builtin_nontemporal_load(mrowp + 16);
    }

    // ---- Online softmax over these 32 keys (max via DPP butterflies). ----
    float corr[8];
#pragma unroll
    for (int r = 0; r < 8; ++r) {
      float v = red_max16(fmaxf(s0[r], s1[r]));
      float mn = fmaxf(mrow[r], v);
      corr[r] = __expf(mrow[r] - mn);
      mrow[r] = mn;
    }

    // Exponentiate, accumulate per-lane partial denominators, stage P.
#pragma unroll
    for (int r = 0; r < 8; ++r) {
      float p0 = __expf(s0[r] - mrow[r]);
      float p1 = __expf(s1[r] - mrow[r]);
      p_lds[wave][r + 8 * half][col]      = (_Float16)p0;
      p_lds[wave][r + 8 * half][16 + col] = (_Float16)p1;
      lsum[r] = lsum[r] * corr[r] + p0 + p1;   // no cross-lane work here
    }
    // p_lds is private to this wave and DS ops are in-order per wave; a
    // wavefront-scope fence just stops compiler reordering (no barrier).
    __builtin_amdgcn_fence(__ATOMIC_ACQ_REL, "wavefront");

    // Re-read P as a 16x32 A-fragment (row M per lane).
    v16h ap = load_frag16(&p_lds[wave][0][0], 32, col, 0, half);

    // ---- acc = diag(corr) * acc + P V, V B-fragments pipelined ahead. ----
    const _Float16* vb = &vbuf[cur][0][0];
    v16h bv = load_frag16(vb, 32, col, 0, half);
#pragma unroll
    for (int j = 0; j < 8; ++j) {
      v16h nv;
      if (j < 7) nv = load_frag16(vb, 32, 16 * (j + 1) + col, 0, half);
      v8f a = acc[j];
#pragma unroll
      for (int r = 0; r < 8; ++r) a[r] *= corr[r];
      acc[j] = __builtin_amdgcn_wmma_f32_16x16x32_f16(
          false, ap, false, bv, (short)0, a, false, false);
      if (j < 7) bv = nv;
    }
    __syncthreads();   // all reads of buffer `cur` done before it is refilled
  }

  // ---- Epilogue: reduce the deferred denominators once, then scale. ----
  float rinv[8];
#pragma unroll
  for (int r = 0; r < 8; ++r)
    rinv[r] = __builtin_amdgcn_rcpf(red_sum16(lsum[r]));

  float* obase = out + ((size_t)b * SQ + q0) * D;
#pragma unroll
  for (int j = 0; j < 8; ++j)
#pragma unroll
    for (int r = 0; r < 8; ++r)
      __builtin_nontemporal_store(
          acc[j][r] * rinv[r],
          &obase[(size_t)(r + 8 * half) * D + 16 * j + col]);
}

// ---------------------------------------------------------------------------
extern "C" void kernel_launch(void* const* d_in, const int* in_sizes, int n_in,
                              void* d_out, int out_size, void* d_ws,
                              size_t ws_size, hipStream_t stream) {
  const float* q    = (const float*)d_in[0];
  const float* k    = (const float*)d_in[1];
  const float* v    = (const float*)d_in[2];
  const float* mask = (const float*)d_in[3];
  float* out = (float*)d_out;

  _Float16* qh = (_Float16*)d_ws;                 // 4 MB
  _Float16* kh = qh + (size_t)B * SQ * D;         // 4 MB
  _Float16* vt = kh + (size_t)B * SK * D;         // 4 MB  (12 MB total)

  const int n = B * SQ * D;
  convert_qk_kernel<<<(n + 255) / 256, 256, 0, stream>>>(q, k, qh, kh, n);

  transpose_v_kernel<<<dim3(SK / 32, D / 32, B), dim3(32, 8), 0, stream>>>(v, vt);

  const int tiles = B * (SQ / 16);                // 1024 wave-tiles
  flash_attn_kernel<<<tiles / 4, 128, 0, stream>>>(mask, qh, kh, vt, out);
}
